// KWayPred_37812892074614
// MI455X (gfx1250) — compile-verified, matrix operand order checked by reading, not run on hardware
//
#include <hip/hip_runtime.h>

// ---------------------------------------------------------------------------
// out[n, r] = h_u[n] @ W_r @ h_v[n]    N=32768, DIM=256, R=65, f32 reference
//
// MI455X / gfx1250 (wave32) strategy:
//   * ~2.8e11 FLOPs over ~90 MB -> AI ~3100 FLOP/B: pure matrix-core bound.
//   * bf16x3 split-precision GEMM on v_wmma_f32_16x16x32_bf16
//     (hi*hi + hi*lo + lo*hi into one f32 accumulator): ~f32 accuracy at
//     ~2.7x the FLOP rate of native V_WMMA_F32_16X16X4_F32.
//   * Prep kernels write hi/lo bf16 copies of h_u and transposed W into d_ws.
//   * Main kernel: WG = 128 rows x 1 edge-type, 8 waves. A-tile resident in
//     LDS; W_r streamed in DOUBLE-BUFFERED 32-col panels staged with
//     GLOBAL_LOAD_ASYNC_TO_LDS_B128 (ASYNCcnt) overlapping the WMMAs.
//     Two interleaved accumulator chains per wave for WMMA ILP.
//     Exact f32 epilogue dot with h_v + shfl_xor lane reduction.
// ---------------------------------------------------------------------------

typedef unsigned short u16;
typedef __attribute__((ext_vector_type(16))) __bf16 v16bf;
typedef __attribute__((ext_vector_type(8)))  float  v8f;
typedef __attribute__((ext_vector_type(4)))  unsigned int u32x4;
// Exact pointee type of the async-LDS builtin's pointer params (per hipcc diag):
typedef int v4i __attribute__((__vector_size__(16)));
typedef __attribute__((address_space(1))) v4i gv4i;   // global
typedef __attribute__((address_space(3))) v4i lv4i;   // LDS

#define DIMD    256
#define NEDGE   65
#define NROWS   32768
#define M_TILE  128
#define E_PANEL 32
#define NPANEL  (DIMD / E_PANEL)   // 8
#define LDSD    264                // padded bf16 row stride (multiple of 8; kills bank conflicts)
#define KSTEPS  8                  // 256 / 32

#if __has_builtin(__builtin_amdgcn_global_load_async_to_lds_b128)
#define HAVE_ASYNC_LDS 1
#endif

// ---- 16B global -> LDS copy: async (ASYNCcnt path) if available -----------
__device__ __forceinline__ void copy16B(const u16* __restrict__ g, u16* l) {
#ifdef HAVE_ASYNC_LDS
    __builtin_amdgcn_global_load_async_to_lds_b128((gv4i*)g, (lv4i*)l, 0, 0);
#else
    *(u32x4*)l = *(const u32x4*)g;
#endif
}

__device__ __forceinline__ void async_fence() {
#ifdef HAVE_ASYNC_LDS
#if __has_builtin(__builtin_amdgcn_s_wait_asynccnt)
    __builtin_amdgcn_s_wait_asynccnt(0);
#else
    asm volatile("s_wait_asynccnt 0" ::: "memory");
#endif
#endif
}

// hi = bit-truncated bf16 (exact prefix of x), lo = RN bf16 of remainder.
__device__ __forceinline__ void split_bf16(float x, u16& hi, u16& lo) {
    union { float f; unsigned u; } a; a.f = x;
    hi = (u16)(a.u >> 16);
    union { float f; unsigned u; } h; h.u = a.u & 0xFFFF0000u;
    union { float f; unsigned u; } r; r.f = x - h.f;
    lo = (u16)((r.u + 0x8000u) >> 16);
}

__global__ void k_split_hu(const float* __restrict__ src,
                           u16* __restrict__ hi, u16* __restrict__ lo, int n) {
    int i = blockIdx.x * blockDim.x + threadIdx.x;
    if (i < n) split_bf16(src[i], hi[i], lo[i]);
}

// Wt[r][e][d] = W[r][d][e], split into hi/lo bf16.
__global__ void k_transpose_w(const float* __restrict__ W,
                              u16* __restrict__ hi, u16* __restrict__ lo) {
    int i = blockIdx.x * blockDim.x + threadIdx.x;   // over R*DIM*DIM, exact
    int d = i & (DIMD - 1);
    int e = (i >> 8) & (DIMD - 1);
    int r = i >> 16;
    float x = W[((size_t)r << 16) + ((size_t)d << 8) + (size_t)e];
    split_bf16(x, hi[i], lo[i]);
}

// 16x32 bf16 A/B fragment per documented layout: lane L -> row/col = L%16,
// K-base = 8*(L/16); two contiguous 16B LDS chunks at K-base and K-base+16.
__device__ __forceinline__ v16bf ld_frag(const u16* p) {
    union { v16bf v; u32x4 q[2]; } f;
    f.q[0] = *(const u32x4*)(p);
    f.q[1] = *(const u32x4*)(p + 16);
    return f.v;
}

__device__ __forceinline__ void stage_b(const u16* __restrict__ Wh,
                                        const u16* __restrict__ Wl,
                                        u16* dh, u16* dl, int ep, int tid) {
    for (int c = tid; c < E_PANEL * DIMD / 8; c += 256) {   // 1024 chunks, 4/thread
        int row = c >> 5;
        int col = (c & 31) << 3;
        size_t g = (size_t)(ep * E_PANEL + row) * DIMD + col;
        copy16B(Wh + g, dh + row * LDSD + col);
        copy16B(Wl + g, dl + row * LDSD + col);
    }
}

__global__ __launch_bounds__(256, 1)
void k_bilinear(const float* __restrict__ hv,
                const u16* __restrict__ hu_hi, const u16* __restrict__ hu_lo,
                const u16* __restrict__ wt_hi, const u16* __restrict__ wt_lo,
                float* __restrict__ out) {
    extern __shared__ u16 smem[];
    u16* sAh = smem;                          // 128*264 bf16 hi
    u16* sAl = sAh + M_TILE * LDSD;           // 128*264 bf16 lo
    u16* sB  = sAl + M_TILE * LDSD;           // 2 x (hi 32*264 + lo 32*264)
    const int BBUF = E_PANEL * LDSD * 2;      // u16 per double-buffer slot

    const int tid   = threadIdx.x;
    const int lane  = tid & 31;
    const int wave  = tid >> 5;
    const int l15   = lane & 15;
    const int lhalf = lane >> 4;
    const int n0    = blockIdx.x * M_TILE;
    const int r     = blockIdx.y;

    const u16* Wh = wt_hi + ((size_t)r << 16);
    const u16* Wl = wt_lo + ((size_t)r << 16);

    // Stage the A tile (h_u rows n0..n0+127, hi & lo) into LDS.
    for (int c = tid; c < M_TILE * DIMD / 8; c += 256) {    // 4096 chunks, 16/thread
        int row = c >> 5;
        int col = (c & 31) << 3;
        size_t g = (size_t)(n0 + row) * DIMD + col;
        copy16B(hu_hi + g, sAh + row * LDSD + col);
        copy16B(hu_lo + g, sAl + row * LDSD + col);
    }
    // Prologue: stage W panel 0 into buffer 0.
    stage_b(Wh, Wl, sB, sB + E_PANEL * LDSD, 0, tid);
    async_fence();
    __syncthreads();

    float vacc[8];
#pragma unroll
    for (int i = 0; i < 8; ++i) vacc[i] = 0.f;

    const int arow = wave * 16 + l15;          // this wave's 16-row m-tile
    const u16* pAh = sAh + arow * LDSD;
    const u16* pAl = sAl + arow * LDSD;
    const int nbase = n0 + wave * 16 + lhalf * 8;

    for (int ep = 0; ep < NPANEL; ++ep) {
        u16* cur = sB + (ep & 1) * BBUF;
        const u16* cBh = cur;
        const u16* cBl = cur + E_PANEL * LDSD;

        // Software pipeline: issue next panel's fill into the other buffer
        // (async loads overlap with the WMMAs below).
        if (ep + 1 < NPANEL) {
            u16* nxt = sB + ((ep + 1) & 1) * BBUF;
            stage_b(Wh, Wl, nxt, nxt + E_PANEL * LDSD, ep + 1, tid);
        }

        // Two interleaved 16x16 accumulator chains (e-tiles et=0,1).
        const u16* pB0h = cBh + l15 * LDSD;
        const u16* pB1h = cBh + (16 + l15) * LDSD;
        const u16* pB0l = cBl + l15 * LDSD;
        const u16* pB1l = cBl + (16 + l15) * LDSD;
        v8f acc0 = {}, acc1 = {};
#pragma unroll
        for (int ks = 0; ks < KSTEPS; ++ks) {
            const int k0 = ks * 32 + lhalf * 8;
            v16bf ah  = ld_frag(pAh + k0);
            v16bf al  = ld_frag(pAl + k0);
            v16bf b0h = ld_frag(pB0h + k0);
            v16bf b1h = ld_frag(pB1h + k0);
            v16bf b0l = ld_frag(pB0l + k0);
            v16bf b1l = ld_frag(pB1l + k0);
            acc0 = __builtin_amdgcn_wmma_f32_16x16x32_bf16(
                       false, ah, false, b0h, (short)0, acc0, false, false);
            acc1 = __builtin_amdgcn_wmma_f32_16x16x32_bf16(
                       false, ah, false, b1h, (short)0, acc1, false, false);
            acc0 = __builtin_amdgcn_wmma_f32_16x16x32_bf16(
                       false, ah, false, b0l, (short)0, acc0, false, false);
            acc1 = __builtin_amdgcn_wmma_f32_16x16x32_bf16(
                       false, ah, false, b1l, (short)0, acc1, false, false);
            acc0 = __builtin_amdgcn_wmma_f32_16x16x32_bf16(
                       false, al, false, b0h, (short)0, acc0, false, false);
            acc1 = __builtin_amdgcn_wmma_f32_16x16x32_bf16(
                       false, al, false, b1h, (short)0, acc1, false, false);
        }

        // Exact f32 epilogue: fold both 16x16 C-tiles against h_v.
        // C layout: vgpr i, lane L -> row M = i + 8*(L/16), col N = L%16.
        const int ecol0 = ep * E_PANEL + l15;
#pragma unroll
        for (int i = 0; i < 8; ++i) {
            const float* hvr = hv + (size_t)(nbase + i) * DIMD + ecol0;
            vacc[i] += acc0[i] * hvr[0] + acc1[i] * hvr[16];
        }

        async_fence();     // next panel's LDS writes landed
        __syncthreads();   // everyone done reading current panel
    }

    // Reduce over the 16 e-columns held by lanes of each half-wave.
#pragma unroll
    for (int i = 0; i < 8; ++i) {
        float v = vacc[i];
        v += __shfl_xor(v, 1);
        v += __shfl_xor(v, 2);
        v += __shfl_xor(v, 4);
        v += __shfl_xor(v, 8);
        if (l15 == 0) out[(size_t)(nbase + i) * NEDGE + r] = v;
    }
}

extern "C" void kernel_launch(void* const* d_in, const int* in_sizes, int n_in,
                              void* d_out, int out_size, void* d_ws, size_t ws_size,
                              hipStream_t stream) {
    const float* h_u = (const float*)d_in[0];
    const float* h_v = (const float*)d_in[1];
    const float* W   = (const float*)d_in[2];
    float* out = (float*)d_out;

    // Workspace layout (bf16 halves), ~50.6 MB total.
    u16* ws16  = (u16*)d_ws;
    const size_t huN = (size_t)NROWS * DIMD;          // 8,388,608
    const size_t wN  = (size_t)NEDGE * DIMD * DIMD;   // 4,259,840
    u16* hu_hi = ws16;
    u16* hu_lo = hu_hi + huN;
    u16* wt_hi = hu_lo + huN;
    u16* wt_lo = wt_hi + wN;

    // Prep: split h_u into bf16 hi/lo; transpose+split W.
    k_split_hu<<<(unsigned)((huN + 255) / 256), 256, 0, stream>>>(h_u, hu_hi, hu_lo, (int)huN);
    k_transpose_w<<<(unsigned)(wN / 256), 256, 0, stream>>>(W, wt_hi, wt_lo);

    // Main bf16x3 WMMA kernel. LDS: A(hi+lo) + 2x W-panel(hi+lo) = 202,752 B.
    const int smem_bytes =
        (M_TILE * LDSD * 2 + 2 * E_PANEL * LDSD * 2) * (int)sizeof(u16);
    (void)hipFuncSetAttribute((const void*)k_bilinear,
                              hipFuncAttributeMaxDynamicSharedMemorySize, smem_bytes);
    dim3 grid(NROWS / M_TILE, NEDGE);
    k_bilinear<<<grid, 256, smem_bytes, stream>>>(h_v, hu_hi, hu_lo, wt_hi, wt_lo, out);
}
